// Head_69930657513857
// MI455X (gfx1250) — matrix operand order repaired and checked by reading, not compiled, not run on hardware
//
#include <hip/hip_runtime.h>

typedef _Float16 v16h __attribute__((ext_vector_type(16)));
typedef float    v8f  __attribute__((ext_vector_type(8)));

#define B_   8
#define L_   2048
#define HS_  128
#define DEC_ 4096
#define BM_  128
#define BN_  64
#define NW_  8      // waves per block
#define NT_  256    // threads per block
#define NKB_ (L_ / BN_)   // 32 tiles of real (non-cache) keys

// exp2-domain score scale: log2(e) / sqrt(HS)
#define KSCALE_ (1.44269504088896340736f * 0.08838834764831844055f)

// max with a DPP16 row-rotate of the other operand (rotation butterfly step).
// ROW_ROR:n encoding = 0x120 | n ; rows are 16 lanes wide == reduction domain.
template <int CTRL>
__device__ __forceinline__ float rotmax16(float x) {
  int xi = __float_as_int(x);
  int yi = __builtin_amdgcn_update_dpp(xi, xi, CTRL, 0xF, 0xF, true);
  return __builtin_fmaxf(x, __int_as_float(yi));
}

__global__ __launch_bounds__(NT_) void fa_ones_cache_kernel(
    const float* __restrict__ K, const float* __restrict__ Q,
    const float* __restrict__ V, float* __restrict__ O)
{
  __shared__ __align__(32) _Float16 sK [BN_ * HS_];          // [key][dim], row-major
  __shared__ __align__(32) _Float16 sVt[HS_ * BN_];          // [dim][key], transposed
  __shared__ __align__(32) _Float16 sP [NW_ * 2 * 32 * 16];  // per-wave P in A-layout blobs

  const int tid   = threadIdx.x;
  const int wave  = tid >> 5;
  const int lane  = tid & 31;
  const int lhalf = lane >> 4;
  const int lm    = lane & 15;

  const int b     = blockIdx.x / (L_ / BM_);
  const int qblk  = blockIdx.x % (L_ / BM_);
  const int qrow0 = qblk * BM_ + wave * 16;

  // ---- load Q as resident WMMA A-fragments (f16, pre-scaled) + row sums ---
  // A 16x32 frag, contiguous k order used here: lane(&15)=row M; elems 0..15 -> K = 16*(lane>>4)+0..15.
  // Q is scaled by log2(e)/sqrt(HS) BEFORE conversion, so S = Q'K^T is already
  // in the exp2 domain -> no per-iteration scale pass.
  v16h qfrag[4];
  const float* qp = Q + ((size_t)b * L_ + qrow0 + lm) * HS_;
  float rs = 0.f;
#pragma unroll
  for (int c = 0; c < 4; ++c) {
    const int dbase = c * 32 + lhalf * 16;
#pragma unroll
    for (int i = 0; i < 8; ++i) {
      float x0 = qp[dbase + i];
      float x1 = qp[dbase + 8 + i];
      qfrag[c][i]     = (_Float16)(x0 * (float)KSCALE_);
      qfrag[c][8 + i] = (_Float16)(x1 * (float)KSCALE_);
      rs += x0 + x1;
    }
  }
  rs += __shfl_xor(rs, 16, 32);          // combine the two K-halves of the row
  const float s0 = rs * (float)KSCALE_;  // exp2-domain score of every ones-cache key

  // constant ones B-fragment: P x ones accumulates rowsum(P) == softmax denom
  v16h vones;
#pragma unroll
  for (int i = 0; i < 16; ++i) vones[i] = (_Float16)1.0f;

  // ---- online-softmax state, seeded analytically with the 4096 ones keys --
  // C-layout: VGPR r holds row (lhalf*8 + r), lane column N = lm.
  float mrow[8];
  v8f   lacc;          // denominator as a WMMA accumulator tile (lane-replicated)
  v8f   acc[8];
#pragma unroll
  for (int r = 0; r < 8; ++r) {
    mrow[r] = __shfl(s0, lhalf * 8 + r, 32);
    lacc[r] = (float)DEC_;               // 4096 ones-keys, e^0 each
  }
#pragma unroll
  for (int t = 0; t < 8; ++t)
#pragma unroll
    for (int r = 0; r < 8; ++r) acc[t][r] = (float)DEC_;  // 4096 * e^0 * ones

  const float* kbase = K + (size_t)b * L_ * HS_;
  const float* vbase = V + (size_t)b * L_ * HS_;
  _Float16* pw = &sP[wave * (2 * 32 * 16)];

  for (int kb = 0; kb < NKB_; ++kb) {
    // -------- cooperative tile fill: K row-major f16, V transposed f16 -----
    const float4* k4 = (const float4*)(kbase + (size_t)kb * BN_ * HS_);
    const float4* v4 = (const float4*)(vbase + (size_t)kb * BN_ * HS_);
#pragma unroll
    for (int it = 0; it < (BN_ * HS_ / 4) / NT_; ++it) {
      const int i   = it * NT_ + tid;
      const int row = i >> 5;           // == (i*4) >> 7
      const int col = (i * 4) & 127;
      float4 kv = k4[i];
      _Float16* d = &sK[row * HS_ + col];
      d[0] = (_Float16)kv.x; d[1] = (_Float16)kv.y;
      d[2] = (_Float16)kv.z; d[3] = (_Float16)kv.w;
      float4 vv = v4[i];
      sVt[(col + 0) * BN_ + row] = (_Float16)vv.x;
      sVt[(col + 1) * BN_ + row] = (_Float16)vv.y;
      sVt[(col + 2) * BN_ + row] = (_Float16)vv.z;
      sVt[(col + 3) * BN_ + row] = (_Float16)vv.w;
    }
    if (kb + 1 < NKB_) {   // gfx1250 global_prefetch_b8 of next tiles
      const char* nk = (const char*)k4 + BN_ * HS_ * 4;
      const char* nv = (const char*)v4 + BN_ * HS_ * 4;
      __builtin_prefetch(nk + tid * 128, 0, 0);
      __builtin_prefetch(nv + tid * 128, 0, 0);
    }
    __syncthreads();

    // -------- S = Q' K^T (already exp2-scaled): 4 16x16 tiles per wave -----
    // B 32x16 frag, same contiguous k order as A: lane col N = lm, K = 16*(lane>>4)+0..15 -> contiguous
    // 32B from the sK row of key (ct*16+lm).
    v8f S[4];
#pragma unroll
    for (int ct = 0; ct < 4; ++ct) {
      v8f s = {0.f, 0.f, 0.f, 0.f, 0.f, 0.f, 0.f, 0.f};
#pragma unroll
      for (int dc = 0; dc < 4; ++dc) {
        v16h bfrag = *(const v16h*)&sK[(ct * 16 + lm) * HS_ + dc * 32 + lhalf * 16];
        s = __builtin_amdgcn_wmma_f32_16x16x32_f16(false, qfrag[dc], false, bfrag,
                                                   (short)0, s, false, false);
      }
      S[ct] = s;
    }

    // -------- online softmax: row max via DPP16 rotation butterfly ---------
    v8f mx = S[0];
#pragma unroll
    for (int ct = 1; ct < 4; ++ct)
#pragma unroll
      for (int r = 0; r < 8; ++r) mx[r] = __builtin_fmaxf(mx[r], S[ct][r]);
#pragma unroll
    for (int r = 0; r < 8; ++r) {
      float m = mx[r];
      m = rotmax16<0x128>(m);   // ROW_ROR:8
      m = rotmax16<0x124>(m);   // ROW_ROR:4
      m = rotmax16<0x122>(m);   // ROW_ROR:2
      m = rotmax16<0x121>(m);   // ROW_ROR:1
      mx[r] = m;
    }

    float alpha[8];
#pragma unroll
    for (int r = 0; r < 8; ++r) {
      float mn = __builtin_fmaxf(mrow[r], mx[r]);
      alpha[r] = __builtin_exp2f(mrow[r] - mn);
      mrow[r]  = mn;
    }
#pragma unroll
    for (int ct = 0; ct < 4; ++ct)
#pragma unroll
      for (int r = 0; r < 8; ++r)
        S[ct][r] = __builtin_exp2f(S[ct][r] - mrow[r]);

    // rescale O accumulator AND the WMMA-tracked denominator
#pragma unroll
    for (int r = 0; r < 8; ++r) lacc[r] *= alpha[r];
#pragma unroll
    for (int t = 0; t < 8; ++t)
#pragma unroll
      for (int r = 0; r < 8; ++r) acc[t][r] *= alpha[r];

    // -------- P (f16) -> LDS, pre-swizzled into the A-fragment order used here --------
    // Element (row q, key kk64): chunk c=kk64>>5; within chunk, A-lane =
    // q + 16*(kk>>4); byte-slot = kk&15.
#pragma unroll
    for (int ct = 0; ct < 4; ++ct) {
      const int kk64 = ct * 16 + lm;
      const int c    = kk64 >> 5;
      const int kk   = kk64 & 31;
      const int hi   = (kk >> 4) & 1;
      const int pos  = kk & 15;
#pragma unroll
      for (int r = 0; r < 8; ++r) {
        const int qrow = lhalf * 8 + r;
        pw[(c * 32 + (qrow + 16 * hi)) * 16 + pos] = (_Float16)S[ct][r];
      }
    }
    asm volatile("" ::: "memory");   // keep same-wave LDS store->load order in IR

    // -------- O += P V ; l += P x ones  (A-frag = contiguous v16h) ---------
#pragma unroll
    for (int c = 0; c < 2; ++c) {
      v16h pfrag = *(const v16h*)&pw[(c * 32 + lane) * 16];
      lacc = __builtin_amdgcn_wmma_f32_16x16x32_f16(false, pfrag, false, vones,
                                                    (short)0, lacc, false, false);
#pragma unroll
      for (int ot = 0; ot < 8; ++ot) {
        v16h vfrag = *(const v16h*)&sVt[(ot * 16 + lm) * BN_ + c * 32 + lhalf * 16];
        acc[ot] = __builtin_amdgcn_wmma_f32_16x16x32_f16(false, pfrag, false, vfrag,
                                                         (short)0, acc[ot], false, false);
      }
    }
    __syncthreads();
  }

  // ---- epilogue: O = acc / l ----
  float* op = O + ((size_t)b * L_ + qrow0) * HS_;
#pragma unroll
  for (int r = 0; r < 8; ++r) {
    const float inv = 1.0f / lacc[r];
    const int   q   = lhalf * 8 + r;
#pragma unroll
    for (int ot = 0; ot < 8; ++ot)
      op[q * HS_ + ot * 16 + lm] = acc[ot][r] * inv;
  }
}

extern "C" void kernel_launch(void* const* d_in, const int* in_sizes, int n_in,
                              void* d_out, int out_size, void* d_ws, size_t ws_size,
                              hipStream_t stream) {
  (void)in_sizes; (void)n_in; (void)d_ws; (void)ws_size; (void)out_size;
  const float* K = (const float*)d_in[0];   // key_token
  const float* Q = (const float*)d_in[1];   // query_token
  const float* V = (const float*)d_in[2];   // value_token
  float* O = (float*)d_out;
  dim3 grid(B_ * (L_ / BM_));   // 128 blocks
  dim3 block(NT_);              // 8 waves (wave32)
  fa_ones_cache_kernel<<<grid, block, 0, stream>>>(K, Q, V, O);
}
